// Attention_block_32349693673648
// MI455X (gfx1250) — matrix operand order repaired and checked by reading, not compiled, not run on hardware
//
#include <hip/hip_runtime.h>
#include <hip/hip_bf16.h>

#define NN 50000
#define EE 1600000
#define DD 128       // in/out feature dim (H*C)
#define HH 4
#define CC 32
#define EDD 64
#define NEG_SLOPE 0.2f
#define NTILES (NN / 16)   // 3125 exact

typedef __attribute__((ext_vector_type(16))) _Float16 v16h;
typedef __attribute__((ext_vector_type(8)))  float    v8f;

__device__ __forceinline__ void atomicMaxF(float* addr, float val) {
    // valid total-order float max via sign-split integer atomics (init must be finite)
    if (val >= 0.0f) atomicMax((int*)addr, __float_as_int(val));
    else             atomicMin((unsigned int*)addr, __float_as_uint(val));
}

// ---------------- init kernels ----------------
__global__ void init_scratch_kernel(float* __restrict__ deg, float* __restrict__ loop_sum,
                                    float* __restrict__ m, float* __restrict__ s) {
    int i = blockIdx.x * blockDim.x + threadIdx.x;
    if (i < NN * EDD) loop_sum[i] = 0.0f;
    if (i < NN)       deg[i] = 0.0f;
    if (i < NN * HH)  { m[i] = -1e30f; s[i] = 0.0f; }
}

__global__ void init_out_kernel(const float* __restrict__ bias, float* __restrict__ out) {
    int g = blockIdx.x * blockDim.x + threadIdx.x;
    if (g < NN * DD) out[g] = bias[g & (DD - 1)];
}

// v[d][h] = sum_c We[d, h*C+c] * att_edge[h, c]   (folds edge projection into a 64x4 matrix)
__global__ void vedge_kernel(const float* __restrict__ We, const float* __restrict__ att_edge,
                             float* __restrict__ v) {
    int t = threadIdx.x;              // 256 threads = 64 dims x 4 heads
    int d = t >> 2, h = t & 3;
    float s = 0.0f;
    #pragma unroll
    for (int c = 0; c < CC; ++c) s += We[d * DD + h * CC + c] * att_edge[h * CC + c];
    v[d * HH + h] = s;
}

// ---------------- WMMA node projection: xp = x @ W (f16 in, f32 acc) ----------------
__global__ void __launch_bounds__(256)
proj_wmma_kernel(const float* __restrict__ x, const float* __restrict__ W,
                 float* __restrict__ xp) {
    // W transposed into LDS as f16: Wt[col][k], row stride 144 halfs (288B, 32B-aligned rows)
    __shared__ _Float16 Wt[DD][DD + 16];
    int tid = threadIdx.x;
    for (int i = tid; i < DD * DD; i += 256) {
        int k = i >> 7, c = i & (DD - 1);
        Wt[c][k] = (_Float16)W[i];
    }
    __syncthreads();

    int wave = tid >> 5;
    int lane = tid & 31;
    int tile = blockIdx.x * 8 + wave;          // one 16-row tile per wave
    if (tile >= NTILES) return;                // wave-uniform: EXEC stays all-ones
    int rowBase = tile * 16;
    int r = rowBase + (lane & 15);
    int khalf = lane >> 4;                     // 0: lanes 0-15, 1: lanes 16-31
    const float* xrow = x + (size_t)r * DD;

    v8f zero = {};
    v8f acc[8];
    #pragma unroll
    for (int t = 0; t < 8; ++t) acc[t] = zero;

    #pragma unroll
    for (int ks = 0; ks < 4; ++ks) {           // K = 128 in 4 steps of 32
        int kbase = ks * 32;
        // A (16x32 f16) layout: lane<16 -> K = kbase+0..15,
        //                       lane>=16 -> K = kbase+16..31
        int b1 = kbase + khalf * 16;
        int b2 = b1 + 8;
        float4 f0 = *(const float4*)(xrow + b1);
        float4 f1 = *(const float4*)(xrow + b1 + 4);
        float4 f2 = *(const float4*)(xrow + b2);
        float4 f3 = *(const float4*)(xrow + b2 + 4);
        v16h a;
        a[0]=(_Float16)f0.x;  a[1]=(_Float16)f0.y;  a[2]=(_Float16)f0.z;  a[3]=(_Float16)f0.w;
        a[4]=(_Float16)f1.x;  a[5]=(_Float16)f1.y;  a[6]=(_Float16)f1.z;  a[7]=(_Float16)f1.w;
        a[8]=(_Float16)f2.x;  a[9]=(_Float16)f2.y;  a[10]=(_Float16)f2.z; a[11]=(_Float16)f2.w;
        a[12]=(_Float16)f3.x; a[13]=(_Float16)f3.y; a[14]=(_Float16)f3.z; a[15]=(_Float16)f3.w;

        // B (32x16 f16): lane = column (mod 16); lane<16 -> K=kbase+0..15, lane>=16 -> K=kbase+16..31
        int koff = kbase + khalf * 16;
        #pragma unroll
        for (int nt = 0; nt < 8; ++nt) {
            int col = nt * 16 + (lane & 15);
            v16h b = *reinterpret_cast<const v16h*>(&Wt[col][koff]);  // 32B-aligned DS load
            acc[nt] = __builtin_amdgcn_wmma_f32_16x16x32_f16(
                false, a, false, b, (short)0, acc[nt], false, false);
        }
    }

    // C/D layout: VGPR v -> row M=v (lanes 0-15) / M=v+8 (lanes 16-31), col N = lane%16
    int colL = lane & 15;
    int rOut = rowBase + khalf * 8;
    #pragma unroll
    for (int nt = 0; nt < 8; ++nt) {
        #pragma unroll
        for (int v = 0; v < 8; ++v) {
            xp[(size_t)(rOut + v) * DD + nt * 16 + colL] = acc[nt][v];
        }
    }
}

// ---------------- per-node attention coefficients ----------------
__global__ void asrc_adst_kernel(const float* __restrict__ xp,
                                 const float* __restrict__ att_src, const float* __restrict__ att_dst,
                                 float* __restrict__ a_src, float* __restrict__ a_dst) {
    int i = blockIdx.x * blockDim.x + threadIdx.x;
    if (i >= NN * HH) return;
    int n = i >> 2, h = i & 3;
    const float* xr = xp + (size_t)n * DD + h * CC;
    const float* as = att_src + h * CC;
    const float* ad = att_dst + h * CC;
    float s1 = 0.0f, s2 = 0.0f;
    #pragma unroll
    for (int c = 0; c < CC; ++c) { float xv = xr[c]; s1 += xv * as[c]; s2 += xv * ad[c]; }
    a_src[i] = s1; a_dst[i] = s2;
}

// ---------------- edge pass 1: degree + loop_attr numerator ----------------
__global__ void edge_accum_kernel(const int* __restrict__ dst, const float* __restrict__ edge_attr,
                                  float* __restrict__ deg, float* __restrict__ loop_sum) {
    int t = threadIdx.x;                       // 256 = 4 edges x 64 dims
    int e = blockIdx.x * 4 + (t >> 6);
    if (e >= EE) return;
    int d = t & 63;
    int dn = dst[e];
    atomicAdd(&loop_sum[(size_t)dn * EDD + d], edge_attr[(size_t)e * EDD + d]);
    if (d == 0) atomicAdd(&deg[dn], 1.0f);
}

// ---------------- edge pass 2: alpha (leaky) + segment max ----------------
__global__ void alpha_edge_kernel(const int* __restrict__ src, const int* __restrict__ dst,
                                  const float* __restrict__ edge_attr, const float* __restrict__ vedge,
                                  const float* __restrict__ a_src, const float* __restrict__ a_dst,
                                  float* __restrict__ alpha, float* __restrict__ m) {
    __shared__ float vs[EDD * HH];
    int t = threadIdx.x;
    if (t < EDD * HH) vs[t] = vedge[t];
    __syncthreads();
    int e = blockIdx.x * blockDim.x + t;
    if (e >= EE) return;
    const float* ea = edge_attr + (size_t)e * EDD;
    float a0 = 0, a1 = 0, a2 = 0, a3 = 0;
    #pragma unroll 8
    for (int d = 0; d < EDD; ++d) {
        float xv = ea[d];
        a0 += xv * vs[d * 4 + 0]; a1 += xv * vs[d * 4 + 1];
        a2 += xv * vs[d * 4 + 2]; a3 += xv * vs[d * 4 + 3];
    }
    int sn = src[e], dn = dst[e];
    float al[4] = {a0, a1, a2, a3};
    #pragma unroll
    for (int h = 0; h < 4; ++h) {
        float a = al[h] + a_src[sn * 4 + h] + a_dst[dn * 4 + h];
        a = (a >= 0.0f) ? a : NEG_SLOPE * a;
        alpha[(size_t)e * 4 + h] = a;
        atomicMaxF(&m[dn * 4 + h], a);
    }
}

// self-loop alpha for node n (edge index E+n), edge_attr = mean of incoming attrs
__global__ void alpha_self_kernel(const float* __restrict__ loop_sum, const float* __restrict__ deg,
                                  const float* __restrict__ vedge,
                                  const float* __restrict__ a_src, const float* __restrict__ a_dst,
                                  float* __restrict__ alpha, float* __restrict__ m) {
    __shared__ float vs[EDD * HH];
    int t = threadIdx.x;
    if (t < EDD * HH) vs[t] = vedge[t];
    __syncthreads();
    int n = blockIdx.x * blockDim.x + t;
    if (n >= NN) return;
    float inv = 1.0f / fmaxf(deg[n], 1.0f);
    const float* ls = loop_sum + (size_t)n * EDD;
    float a0 = 0, a1 = 0, a2 = 0, a3 = 0;
    #pragma unroll 8
    for (int d = 0; d < EDD; ++d) {
        float la = ls[d] * inv;
        a0 += la * vs[d * 4 + 0]; a1 += la * vs[d * 4 + 1];
        a2 += la * vs[d * 4 + 2]; a3 += la * vs[d * 4 + 3];
    }
    float al[4] = {a0, a1, a2, a3};
    #pragma unroll
    for (int h = 0; h < 4; ++h) {
        float a = al[h] + a_src[n * 4 + h] + a_dst[n * 4 + h];
        a = (a >= 0.0f) ? a : NEG_SLOPE * a;
        alpha[(size_t)(EE + n) * 4 + h] = a;
        atomicMaxF(&m[n * 4 + h], a);
    }
}

// ---------------- softmax: exp + segment sum (alpha overwritten with exp) ----------------
__global__ void exp_sum_kernel(const int* __restrict__ dst, float* __restrict__ alpha,
                               const float* __restrict__ m, float* __restrict__ s) {
    int f = blockIdx.x * blockDim.x + threadIdx.x;
    if (f >= EE + NN) return;
    int dn = (f < EE) ? dst[f] : (f - EE);
    #pragma unroll
    for (int h = 0; h < 4; ++h) {
        float ex = expf(alpha[(size_t)f * 4 + h] - m[dn * 4 + h]);
        alpha[(size_t)f * 4 + h] = ex;
        atomicAdd(&s[dn * 4 + h], ex);
    }
}

// ---------------- weighted message scatter ----------------
__global__ void scatter_out_kernel(const int* __restrict__ src, const int* __restrict__ dst,
                                   const float* __restrict__ xp, const float* __restrict__ alpha,
                                   const float* __restrict__ s, float* __restrict__ out) {
    long long g = (long long)blockIdx.x * blockDim.x + threadIdx.x;
    if (g >= (long long)(EE + NN) * DD) return;
    int f = (int)(g >> 7);
    int c = (int)(g & (DD - 1));
    int h = c >> 5;
    int sn, dn;
    if (f < EE) { sn = src[f]; dn = dst[f]; } else { sn = dn = f - EE; }
    float attn = alpha[(size_t)f * 4 + h] / (s[dn * 4 + h] + 1e-16f);
    atomicAdd(&out[(size_t)dn * DD + c], xp[(size_t)sn * DD + c] * attn);
}

extern "C" void kernel_launch(void* const* d_in, const int* in_sizes, int n_in,
                              void* d_out, int out_size, void* d_ws, size_t ws_size,
                              hipStream_t stream) {
    (void)in_sizes; (void)n_in; (void)out_size; (void)ws_size;
    const float* x         = (const float*)d_in[0];
    const int*   eidx      = (const int*)d_in[1];     // [2,E]: src then dst
    const float* edge_attr = (const float*)d_in[2];
    const float* W         = (const float*)d_in[3];
    const float* att_src   = (const float*)d_in[4];
    const float* att_dst   = (const float*)d_in[5];
    const float* We        = (const float*)d_in[6];
    const float* att_edge  = (const float*)d_in[7];
    const float* bias      = (const float*)d_in[8];
    float* out = (float*)d_out;
    const int* srcI = eidx;
    const int* dstI = eidx + EE;

    float* ws = (float*)d_ws;
    float* xp       = ws;  ws += (size_t)NN * DD;
    float* a_src    = ws;  ws += NN * HH;
    float* a_dst    = ws;  ws += NN * HH;
    float* vedge    = ws;  ws += EDD * HH;
    float* deg      = ws;  ws += NN;
    float* loop_sum = ws;  ws += (size_t)NN * EDD;
    float* m        = ws;  ws += NN * HH;
    float* sden     = ws;  ws += NN * HH;
    float* alpha    = ws;  ws += (size_t)(EE + NN) * HH;

    init_scratch_kernel<<<(NN * EDD + 255) / 256, 256, 0, stream>>>(deg, loop_sum, m, sden);
    init_out_kernel<<<(NN * DD + 255) / 256, 256, 0, stream>>>(bias, out);
    vedge_kernel<<<1, 256, 0, stream>>>(We, att_edge, vedge);
    proj_wmma_kernel<<<(NTILES + 7) / 8, 256, 0, stream>>>(x, W, xp);
    asrc_adst_kernel<<<(NN * HH + 255) / 256, 256, 0, stream>>>(xp, att_src, att_dst, a_src, a_dst);
    edge_accum_kernel<<<(EE + 3) / 4, 256, 0, stream>>>(dstI, edge_attr, deg, loop_sum);
    alpha_edge_kernel<<<(EE + 255) / 256, 256, 0, stream>>>(srcI, dstI, edge_attr, vedge,
                                                            a_src, a_dst, alpha, m);
    alpha_self_kernel<<<(NN + 255) / 256, 256, 0, stream>>>(loop_sum, deg, vedge,
                                                            a_src, a_dst, alpha, m);
    exp_sum_kernel<<<(EE + NN + 255) / 256, 256, 0, stream>>>(dstI, alpha, m, sden);
    long long tot = (long long)(EE + NN) * DD;
    scatter_out_kernel<<<(unsigned)((tot + 255) / 256), 256, 0, stream>>>(srcI, dstI, xp,
                                                                          alpha, sden, out);
}
